// HyperbolicLinearAdvanced_48808008351953
// MI455X (gfx1250) — matrix operand order repaired and checked by reading, not compiled, not run on hardware
//
#include <hip/hip_runtime.h>
#include <hip/hip_bf16.h>
#include <math.h>

#define EPSF 1e-6f

typedef __attribute__((ext_vector_type(16))) __bf16 v16bf;
typedef __attribute__((ext_vector_type(8)))  float  v8f;
typedef __attribute__((ext_vector_type(4)))  unsigned int u32x4;
typedef __attribute__((ext_vector_type(8)))  int          i32x8;
typedef __attribute__((ext_vector_type(4)))  int          i32x4;

#if __has_builtin(__builtin_amdgcn_tensor_load_to_lds)
#define ATHENA_HAS_TDM 1
#endif

#ifdef ATHENA_HAS_TDM
#define STAGE_ASYNC_N 1   // per-thread async instrs per stage (A only; B via TDM)
#else
#define STAGE_ASYNC_N 5   // 1 A chunk + 4 B chunks per thread
#endif

#if __has_builtin(__builtin_amdgcn_s_wait_asynccnt)
#define WAIT_ASYNCCNT(n) __builtin_amdgcn_s_wait_asynccnt(n)
#else
#define WAIT_ASYNCCNT(n) asm volatile("s_wait_asynccnt %0" :: "i"(n) : "memory")
#endif

__device__ __forceinline__ unsigned int lds_off(const void* p) {
    // generic shared pointer: low 32 bits are the byte offset in the WG's LDS
    return (unsigned int)(uintptr_t)p;
}

__device__ __forceinline__ void async_g2l_b128(unsigned int loff, const void* gptr) {
    // LDS[loff..loff+15] = MEM[gptr..gptr+15], tracked with ASYNCcnt
    asm volatile("global_load_async_to_lds_b128 %0, %1, off"
                 :: "v"(loff), "v"(gptr)
                 : "memory");
}

__device__ __forceinline__ float artanh_clip(float v) {
    v = fminf(fmaxf(v, -1.0f + 1e-5f), 1.0f - 1e-5f);
    return 0.5f * logf((1.0f + v) / (1.0f - v));
}

// ---------------------------------------------------------------------------
// Kernel 0: anchor squared norms + softmax of anchor weights (1 block)
// ---------------------------------------------------------------------------
__global__ void __launch_bounds__(256)
prep_kernel(const float* __restrict__ anchors, const float* __restrict__ aw,
            float* __restrict__ a2, float* __restrict__ wsm, int K, int D) {
    __shared__ float sred[8];
    const int lane = threadIdx.x & 31;
    const int wv   = threadIdx.x >> 5;
    for (int k = 0; k < K; ++k) {
        float s = 0.f;
        for (int j = threadIdx.x; j < D; j += blockDim.x) {
            float v = anchors[(size_t)k * D + j];
            s += v * v;
        }
        for (int off = 16; off > 0; off >>= 1) s += __shfl_down(s, off, 32);
        if (lane == 0) sred[wv] = s;
        __syncthreads();
        if (threadIdx.x == 0) {
            float t = 0.f;
            for (int i = 0; i < 8; ++i) t += sred[i];
            a2[k] = t;
        }
        __syncthreads();
    }
    if (threadIdx.x == 0) {
        float mx = -1e30f;
        for (int k = 0; k < K; ++k) mx = fmaxf(mx, aw[k]);
        float s = 0.f;
        for (int k = 0; k < K; ++k) { float e = expf(aw[k] - mx); wsm[k] = e; s += e; }
        for (int k = 0; k < K; ++k) wsm[k] /= s;
    }
}

// ---------------------------------------------------------------------------
// Kernel 1: per-row curvature logit (-> mean softplus via atomicAdd) + ||x||^2
// ---------------------------------------------------------------------------
__global__ void __launch_bounds__(256)
rowstat_kernel(const float* __restrict__ x, const float* __restrict__ cw,
               const float* __restrict__ cb, float* __restrict__ c_acc,
               float* __restrict__ xn2, int D, float invB) {
    const int b = blockIdx.x;
    const float* xr = x + (size_t)b * D;
    float dot = 0.f, ss = 0.f;
    for (int j = threadIdx.x; j < D; j += blockDim.x) {
        float v = xr[j];
        dot += v * cw[j];
        ss  += v * v;
    }
    __shared__ float s1[8], s2[8];
    for (int off = 16; off > 0; off >>= 1) {
        dot += __shfl_down(dot, off, 32);
        ss  += __shfl_down(ss,  off, 32);
    }
    const int lane = threadIdx.x & 31, wv = threadIdx.x >> 5;
    if (lane == 0) { s1[wv] = dot; s2[wv] = ss; }
    __syncthreads();
    if (threadIdx.x == 0) {
        float d = 0.f, s = 0.f;
        for (int i = 0; i < 8; ++i) { d += s1[i]; s += s2[i]; }
        float logit = d + cb[0];
        float sp = fmaxf(logit, 0.f) + log1pf(expf(-fabsf(logit)));  // softplus
        atomicAdd(c_acc, sp * invB);
        xn2[b] = s;
    }
}

// ---------------------------------------------------------------------------
// Kernel 2: z = x @ W^T.
//   Block tile 16(M) x 128(N), K-step 32, 8 waves, one 16x16 WMMA tile each.
//   Per k-step the tiles are staged into LDS (double-buffered):
//     A (16x32 f32)  : global_load_async_to_lds_b128 (ASYNCcnt)
//     B (128x32 f32) : TDM tensor_load_to_lds, 2D D#, LDS padding
//                      (pad 4 dwords every 32 -> LDS row stride 36 floats,
//                       conflict-free 16-lane fragment reads)  (TENSORcnt)
//   Fragments are read from LDS, converted to bf16 hi/lo, and accumulated as
//     z += Ahi*Bhi + Ahi*Blo + Alo*Bhi   via v_wmma_f32_16x16x32_bf16.
// ---------------------------------------------------------------------------
#define LDB 36   // padded LDS row stride in floats (144 B, 16B-aligned rows)

__global__ void __launch_bounds__(256)
wmma_gemm_kernel(const float* __restrict__ x,   // B  x Dk, row-major
                 const float* __restrict__ w,   // Dn x Dk, row-major
                 float* __restrict__ z,         // B  x Dn
                 int Dk, int Dn) {
    __shared__ __align__(16) float As[2][16][LDB];
    __shared__ __align__(16) float Bs[2][128][LDB];

    const int t    = threadIdx.x;
    const int lane = t & 31;
    const int wv   = t >> 5;                    // 0..7
    const int m0   = blockIdx.y * 16;
    const int n0b  = blockIdx.x * 128;
    const int r    = lane & 15;
    const int g    = lane >> 4;

    // A-staging assignment: 128 b128 chunks; threads 128..255 duplicate.
    const int ca   = t & 127;
    const int arow = ca >> 3;
    const int ac4  = (ca & 7) * 4;
    const float* ag_base = x + (size_t)(m0 + arow) * Dk + ac4;

    auto stage = [&](int buf, int k0) {
        // ---- A tile: one async b128 per thread (uniform ASYNCcnt) ----
        async_g2l_b128(lds_off(&As[buf][arow][ac4]), ag_base + k0);
#ifdef ATHENA_HAS_TDM
        // ---- B tile: wave 0 issues one TDM descriptor ----
        if (t < 32) {
            const float* gB = w + (size_t)n0b * Dk + k0;
            unsigned long long ga = (unsigned long long)(uintptr_t)gB;
            u32x4 g0;
            g0[0] = 0x1u;                                   // count=1, load
            g0[1] = lds_off(&Bs[buf][0][0]);                // lds_addr
            g0[2] = (unsigned int)ga;                       // global_addr[31:0]
            g0[3] = (unsigned int)((ga >> 32) & 0x01FFFFFFu) | (2u << 30); // +type=2
            i32x8 g1;
            g1[0] = (int)((2u << 16)        // data_size = 4B
                        | (1u << 20)        // pad_enable
                        | (4u << 22)        // pad_interval: every 32 dwords
                        | (3u << 25));      // pad_amount: 4 dwords -> stride 36
            g1[1] = (int)(((unsigned)Dk & 0xFFFFu) << 16);          // tensor_dim0 lo
            g1[2] = (int)((((unsigned)Dk >> 16) & 0xFFFFu)
                        | (128u << 16));                            // dim0 hi | tensor_dim1 lo
            g1[3] = (int)(32u << 16);                               // dim1 hi=0 | tile_dim0=32
            g1[4] = 128;                                            // tile_dim1=128, tile_dim2=0
            g1[5] = Dk;                                             // tensor_dim0_stride lo32
            g1[6] = 0;                                              // stride hi | dim1_stride lo
            g1[7] = 0;
            i32x4 zg4 = {0, 0, 0, 0};
#if defined(__clang_major__) && __clang_major__ >= 23
            i32x8 zg8 = {0, 0, 0, 0, 0, 0, 0, 0};
            __builtin_amdgcn_tensor_load_to_lds(g0, g1, zg4, zg4, zg8, 0);
#else
            __builtin_amdgcn_tensor_load_to_lds(g0, g1, zg4, zg4, 0);
#endif
        }
#else
        // ---- Fallback: B tile via async b128, 4 chunks per thread ----
        for (int j = 0; j < 4; ++j) {
            int cb   = t + j * 256;                         // 0..1023
            int brow = cb >> 3;
            int bc4  = (cb & 7) * 4;
            async_g2l_b128(lds_off(&Bs[buf][brow][bc4]),
                           w + (size_t)(n0b + brow) * Dk + k0 + bc4);
        }
#endif
    };

    v8f acc = {};
    const int nk = Dk >> 5;
    stage(0, 0);
    for (int i = 0; i < nk; ++i) {
        const int buf  = i & 1;
        const bool more = (i + 1 < nk);
        if (more) stage(buf ^ 1, (i + 1) << 5);
#ifdef ATHENA_HAS_TDM
        if (t < 32) {
            if (more) __builtin_amdgcn_s_wait_tensorcnt(1);
            else      __builtin_amdgcn_s_wait_tensorcnt(0);
        }
#endif
        if (more) { WAIT_ASYNCCNT(STAGE_ASYNC_N); }
        else      { WAIT_ASYNCCNT(0); }
        __syncthreads();

        const float* ap = &As[buf][r][0];
        const float* bp = &Bs[buf][wv * 16 + r][0];
        v16bf ah, al, bh, bl;
#pragma unroll
        for (int q = 0; q < 8; ++q) {
            float f1 = ap[g * 8 + q];       __bf16 h1 = (__bf16)f1;
            float f2 = ap[16 + g * 8 + q];  __bf16 h2 = (__bf16)f2;
            ah[q]     = h1;  al[q]     = (__bf16)(f1 - (float)h1);
            ah[8 + q] = h2;  al[8 + q] = (__bf16)(f2 - (float)h2);
        }
#pragma unroll
        for (int q = 0; q < 16; ++q) {
            float f = bp[(q < 8) ? (g * 8 + q) : (8 + g * 8 + q)];  __bf16 h = (__bf16)f;
            bh[q] = h;  bl[q] = (__bf16)(f - (float)h);
        }
        acc = __builtin_amdgcn_wmma_f32_16x16x32_bf16(false, ah, false, bh,
                                                      (short)0, acc, false, false);
        acc = __builtin_amdgcn_wmma_f32_16x16x32_bf16(false, ah, false, bl,
                                                      (short)0, acc, false, false);
        acc = __builtin_amdgcn_wmma_f32_16x16x32_bf16(false, al, false, bh,
                                                      (short)0, acc, false, false);
        __syncthreads();
    }

    const int n0 = n0b + wv * 16;
#pragma unroll
    for (int q = 0; q < 8; ++q) {
        int m = m0 + g * 8 + q;
        z[(size_t)m * Dn + n0 + r] = acc[q];
    }
}

// ---------------------------------------------------------------------------
// Kernel 3: fused epilogue, one block per row.
//   h = alpha_b * z + bias ; y = beta_b * h
//   Mobius chain collapses to out = S_A*y + sum_k C_k * anchors[k]
//   with scalars from {||y||^2, ||a_k||^2, <y,a_k>}.
// ---------------------------------------------------------------------------
__global__ void __launch_bounds__(256)
epilogue_kernel(const float* __restrict__ z, const float* __restrict__ bias,
                const float* __restrict__ anchors, const float* __restrict__ tvals,
                const float* __restrict__ a2v, const float* __restrict__ wsm,
                const float* __restrict__ c_ptr, const float* __restrict__ xn2,
                float* __restrict__ out, int D, int K) {
    const int b = blockIdx.x;
    const float c  = c_ptr[0];
    const float sc = sqrtf(c);
    const float nx = fmaxf(sqrtf(xn2[b]), EPSF);
    const float alpha = artanh_clip(sc * nx) / (sc * nx);

    const float* zr = z + (size_t)b * D;

    float hh = 0.f, ha[4] = {0.f, 0.f, 0.f, 0.f};
    for (int j = threadIdx.x; j < D; j += blockDim.x) {
        float h = fmaf(alpha, zr[j], bias[j]);
        hh += h * h;
        for (int k = 0; k < K; ++k) ha[k] += h * anchors[(size_t)k * D + j];
    }
    __shared__ float smem[8 * 5];
    __shared__ float bc[5];
    float vals[5];
    vals[0] = hh;
    for (int k = 0; k < 4; ++k) vals[1 + k] = (k < K) ? ha[k] : 0.f;
    for (int off = 16; off > 0; off >>= 1)
        for (int i = 0; i < 5; ++i) vals[i] += __shfl_down(vals[i], off, 32);
    const int lane = threadIdx.x & 31, wv = threadIdx.x >> 5;
    if (lane == 0)
        for (int i = 0; i < 5; ++i) smem[wv * 5 + i] = vals[i];
    __syncthreads();
    if (threadIdx.x == 0)
        for (int i = 0; i < 5; ++i) {
            float tt = 0.f;
            for (int w8 = 0; w8 < 8; ++w8) tt += smem[w8 * 5 + i];
            bc[i] = tt;
        }
    __syncthreads();

    hh = bc[0];
    const float nh   = fmaxf(sqrtf(hh), EPSF);
    const float beta = tanhf(sc * nh) / (sc * nh);
    const float Y2   = beta * beta * hh;

    float SA = 0.f, Ck[4];
    for (int k = 0; k < K; ++k) {
        float YA = beta * bc[1 + k];
        float A2 = a2v[k];
        float cA = 1.f + 2.f * c * (-YA) + c * A2;
        float cB = 1.f - c * Y2;
        float d1 = fmaxf(1.f + 2.f * c * (-YA) + c * c * Y2 * A2, EPSF);
        float p1 = -cA / d1, q1 = cB / d1;
        float nd2 = p1 * p1 * Y2 + 2.f * p1 * q1 * YA + q1 * q1 * A2;
        float nd  = fmaxf(sqrtf(fmaxf(nd2, 0.f)), EPSF);
        float gamma = tanhf(tvals[k] * artanh_clip(sc * nd)) / (sc * nd);
        float p2 = gamma * p1, q2 = gamma * q1;
        float y2m = gamma * gamma * nd2;
        float xym = p2 * Y2 + q2 * YA;
        float cA2 = 1.f + 2.f * c * xym + c * y2m;
        float cB2 = 1.f - c * Y2;
        float d2  = fmaxf(1.f + 2.f * c * xym + c * c * Y2 * y2m, EPSF);
        float Gy = (cA2 + cB2 * p2) / d2;
        float Ga = (cB2 * q2) / d2;
        SA   += wsm[k] * Gy;
        Ck[k] = wsm[k] * Ga;
    }
    const float sy = SA * beta;

    float* outr = out + (size_t)b * D;
    for (int j = threadIdx.x; j < D; j += blockDim.x) {
        float h = fmaf(alpha, zr[j], bias[j]);
        float v = sy * h;
        for (int k = 0; k < K; ++k) v += Ck[k] * anchors[(size_t)k * D + j];
        outr[j] = v;
    }
}

// ---------------------------------------------------------------------------
extern "C" void kernel_launch(void* const* d_in, const int* in_sizes, int n_in,
                              void* d_out, int out_size, void* d_ws, size_t ws_size,
                              hipStream_t stream) {
    const float* x       = (const float*)d_in[0];   // (B, Din)
    const float* wt      = (const float*)d_in[1];   // (Dout, Din)
    const float* bias    = (const float*)d_in[2];   // (Dout,)
    const float* cw      = (const float*)d_in[3];   // (1, Din)
    const float* cb      = (const float*)d_in[4];   // (1,)
    const float* anchors = (const float*)d_in[5];   // (K, Dout)
    const float* tvals   = (const float*)d_in[6];   // (K,)
    const float* aw      = (const float*)d_in[7];   // (K,)

    const int Din  = in_sizes[3];                   // 2048
    const int Dout = in_sizes[2];                   // 2048
    const int Bn   = in_sizes[0] / Din;             // 4096
    const int K    = in_sizes[6];                   // 4

    float* wsf   = (float*)d_ws;
    float* c_acc = wsf + 0;
    float* a2v   = wsf + 8;
    float* wsm   = wsf + 16;
    float* xn2   = wsf + 64;
    size_t zoff  = ((64 + (size_t)Bn) + 63) & ~((size_t)63);
    float* zbuf  = wsf + zoff;

    hipMemsetAsync(c_acc, 0, sizeof(float), stream);
    prep_kernel<<<1, 256, 0, stream>>>(anchors, aw, a2v, wsm, K, Dout);
    rowstat_kernel<<<Bn, 256, 0, stream>>>(x, cw, cb, c_acc, xn2, Din, 1.0f / (float)Bn);
    dim3 gg(Dout / 128, Bn / 16);
    wmma_gemm_kernel<<<gg, 256, 0, stream>>>(x, wt, zbuf, Din, Dout);
    epilogue_kernel<<<Bn, 256, 0, stream>>>(zbuf, bias, anchors, tvals, a2v, wsm,
                                            c_acc, xn2, (float*)d_out, Dout, K);
}
